// DisentangledSelfAttention_42683384988092
// MI455X (gfx1250) — matrix operand order repaired and checked by reading, not compiled, not run on hardware
//
#include <hip/hip_runtime.h>

// ---------------------------------------------------------------------------
// Types for gfx1250 WMMA
// ---------------------------------------------------------------------------
typedef __attribute__((ext_vector_type(16))) __bf16 v16bf;
typedef __attribute__((ext_vector_type(8)))  float  v8f;
typedef __attribute__((ext_vector_type(4)))  unsigned int v4u;
typedef int v4i_vs __attribute__((vector_size(16)));   // matches builtin param type

union BF16Frag  { v16bf bf; v4u q[2]; };
union Stage16   { v4u q[2]; unsigned short us[16]; };

__device__ __forceinline__ unsigned short f2bf(float f) {
  unsigned int u = __float_as_uint(f);
  u += 0x7fffu + ((u >> 16) & 1u);           // round-to-nearest-even
  return (unsigned short)(u >> 16);
}
__device__ __forceinline__ float bf2f(unsigned short b) {
  return __uint_as_float(((unsigned int)b) << 16);
}

// ---------------------------------------------------------------------------
// gfx1250 async global->LDS copy (ASYNCcnt path), with safe fallback
// ---------------------------------------------------------------------------
#if defined(__gfx1250__) && \
    __has_builtin(__builtin_amdgcn_global_load_async_to_lds_b128) && \
    __has_builtin(__builtin_amdgcn_s_wait_asynccnt)
#define HAVE_ASYNC_LDS 1
typedef __attribute__((address_space(1))) v4i_vs* as1_v4i;
typedef __attribute__((address_space(3))) v4i_vs* as3_v4i;
#else
#define HAVE_ASYNC_LDS 0
#endif

// copy 16 bytes (8 bf16) global -> LDS
__device__ __forceinline__ void cp16B(const unsigned short* g, unsigned short* l) {
#if HAVE_ASYNC_LDS
  __builtin_amdgcn_global_load_async_to_lds_b128(
      (as1_v4i)(void*)(const_cast<unsigned short*>(g)),
      (as3_v4i)(void*)(l), 0, 0);
#else
  *(v4u*)(void*)l = *(const v4u*)(const void*)g;
#endif
}
__device__ __forceinline__ void cp_commit() {
#if HAVE_ASYNC_LDS
  __builtin_amdgcn_s_wait_asynccnt(0);
#endif
}

constexpr int NB    = 4;     // batch
constexpr int SEQ   = 1024;  // sequence length
constexpr int DM    = 1024;  // model dim
constexpr int NH    = 16;    // heads
constexpr int HDIM  = 64;    // head dim
constexpr int K2    = 512;   // 2 * buckets
constexpr int SPAN  = 256;
constexpr int MAXREL= 512;

// ---------------------------------------------------------------------------
// fp32 -> bf16 elementwise convert
// ---------------------------------------------------------------------------
__global__ void cvt_bf16_k(const float* __restrict__ s,
                           unsigned short* __restrict__ d, int n) {
  int i = blockIdx.x * 256 + threadIdx.x;
  if (i < n) d[i] = f2bf(s[i]);
}

// ---------------------------------------------------------------------------
// LayerNorm over feature dim of rel_emb [K2, DM] -> bf16
// ---------------------------------------------------------------------------
__global__ __launch_bounds__(256) void ln_bf16_k(const float* __restrict__ x,
                                                 const float* __restrict__ g,
                                                 const float* __restrict__ bb,
                                                 unsigned short* __restrict__ y) {
  __shared__ float red[256];
  int row = blockIdx.x;
  const float* xr = x + (size_t)row * DM;
  float s = 0.f, s2 = 0.f;
  for (int i = threadIdx.x; i < DM; i += 256) { float v = xr[i]; s += v; s2 += v * v; }
  red[threadIdx.x] = s; __syncthreads();
  for (int off = 128; off; off >>= 1) {
    if (threadIdx.x < off) red[threadIdx.x] += red[threadIdx.x + off];
    __syncthreads();
  }
  float mean = red[0] / DM; __syncthreads();
  red[threadIdx.x] = s2; __syncthreads();
  for (int off = 128; off; off >>= 1) {
    if (threadIdx.x < off) red[threadIdx.x] += red[threadIdx.x + off];
    __syncthreads();
  }
  float var = red[0] / DM - mean * mean;
  float inv = rsqrtf(var + 1e-5f);
  for (int i = threadIdx.x; i < DM; i += 256) {
    float v = (xr[i] - mean) * inv * g[i] + bb[i];
    y[(size_t)row * DM + i] = f2bf(v);
  }
}

// ---------------------------------------------------------------------------
// delta -> clipped bucket index table (delta in [-1023,1023] at t = delta+1023)
// ---------------------------------------------------------------------------
__global__ void relidx_k(int* __restrict__ tab) {
  int t = blockIdx.x * 256 + threadIdx.x;
  if (t >= 2048) return;
  int rel = t - 1023;
  if (t == 2047) rel = 1023;  // keep last slot valid
  const int mid = 128;  // BUCKETS/2
  float sgn = rel > 0 ? 1.f : (rel < 0 ? -1.f : 0.f);
  float abs_pos = (rel < mid && rel > -mid) ? (float)(mid - 1) : fabsf((float)rel);
  float log_pos = ceilf(logf(abs_pos / (float)mid) /
                        logf((float)(MAXREL - 1) / (float)mid) * (float)(mid - 1)) + (float)mid;
  float bucket = (abs_pos <= (float)mid) ? (float)rel : log_pos * sgn;
  int bi = (int)bucket + SPAN;
  bi = min(max(bi, 0), 2 * SPAN - 1);
  tab[t] = bi;
}

// ---------------------------------------------------------------------------
// Generic bf16 GEMM: C[M,N] = A[M,K] @ W[K,N] + bias, both row-major bf16.
// 128x128 block tile, 8 waves each 64x32, WMMA 16x16x32 bf16.
// ---------------------------------------------------------------------------
#define GSTR 40   // 32 + 8 pad (80B rows, 16B-aligned)
__global__ __launch_bounds__(256) void gemm_rm_k(
    const unsigned short* __restrict__ A, int lda,
    const unsigned short* __restrict__ W, int ldw,
    const float* __restrict__ bias,
    void* __restrict__ Cout, int ldc,
    int M, int N, int K, int outF32) {
  __shared__ unsigned short Al[128 * GSTR];
  __shared__ unsigned short Wl[128 * GSTR];
  int tid = threadIdx.x;
  int wave = tid >> 5, lane = tid & 31;
  int hh = lane >> 4, n16 = lane & 15;
  int gm0 = blockIdx.y * 128, gn0 = blockIdx.x * 128;
  int wm = (wave >> 2) * 64;   // 0 or 64
  int wn = (wave & 3) * 32;    // 0..96

  // static copy assignments (exact cover, no loops)
  const int ar  = tid >> 1,  ac = (tid & 1) * 16;   // A: 128 rows x 2 halves
  const int wkr = tid >> 3,  wnc = (tid & 7) * 16;  // W: 32 k-rows x 8 n-chunks

  v8f acc[4][2];
  const v8f vzero = {0.f,0.f,0.f,0.f,0.f,0.f,0.f,0.f};
#pragma unroll
  for (int i = 0; i < 4; i++)
#pragma unroll
    for (int j = 0; j < 2; j++) acc[i][j] = vzero;

  for (int kt = 0; kt < K; kt += 32) {
    // A tile 128x32: 16 bf16 per thread, async b128 copies
    {
      const unsigned short* gp = A + (size_t)(gm0 + ar) * lda + kt + ac;
      cp16B(gp,     &Al[ar * GSTR + ac]);
      cp16B(gp + 8, &Al[ar * GSTR + ac + 8]);
    }
    // W tile transposed into Wl[n][k]: stage 16 bf16 in regs, scatter to LDS
    {
      const unsigned short* wp = W + (size_t)(kt + wkr) * ldw + gn0 + wnc;
      Stage16 t;
      t.q[0] = *(const v4u*)(const void*)wp;
      t.q[1] = *(const v4u*)(const void*)(wp + 8);
#pragma unroll
      for (int e = 0; e < 16; e++) Wl[(wnc + e) * GSTR + wkr] = t.us[e];
    }
    if (kt + 32 < K) {  // gfx1250 global_prefetch_b8 of next tiles
      __builtin_prefetch(A + (size_t)(gm0 + ar) * lda + kt + 32, 0, 1);
      __builtin_prefetch(W + (size_t)(kt + 32 + wkr) * ldw + gn0, 0, 1);
    }
    cp_commit();
    __syncthreads();

    BF16Frag af[4], bfg[2];
#pragma unroll
    for (int i = 0; i < 4; i++) {
      const unsigned short* p = &Al[(wm + i * 16 + n16) * GSTR + hh * 8];
      af[i].q[0] = *(const v4u*)(const void*)p;
      af[i].q[1] = *(const v4u*)(const void*)(p + 16);
    }
#pragma unroll
    for (int j = 0; j < 2; j++) {
      const unsigned short* p = &Wl[(wn + j * 16 + n16) * GSTR + hh * 8];
      bfg[j].q[0] = *(const v4u*)(const void*)p;
      bfg[j].q[1] = *(const v4u*)(const void*)(p + 16);
    }
#pragma unroll
    for (int i = 0; i < 4; i++)
#pragma unroll
      for (int j = 0; j < 2; j++)
        acc[i][j] = __builtin_amdgcn_wmma_f32_16x16x32_bf16(
            false, af[i].bf, false, bfg[j].bf, (short)0, acc[i][j], false, false);
    __syncthreads();
  }

#pragma unroll
  for (int i = 0; i < 4; i++) {
    int row = gm0 + wm + i * 16 + hh * 8;
#pragma unroll
    for (int j = 0; j < 2; j++) {
      int col = gn0 + wn + j * 16 + n16;
      float bv = bias ? bias[col] : 0.f;
#pragma unroll
      for (int r = 0; r < 8; r++) {
        float v = acc[i][j][r] + bv;
        if (outF32) ((float*)Cout)[(size_t)(row + r) * ldc + col] = v;
        else ((unsigned short*)Cout)[(size_t)(row + r) * ldc + col] = f2bf(v);
      }
    }
  }
}

// ---------------------------------------------------------------------------
// Batched NT GEMM for c2p/p2c:  C[z][s][n] = X_bh[s,:] . P_h[n,:]  (K = 64)
// X: [NB*SEQ, DM] bf16 head slice, P: [K2, DM] bf16 head slice.
// ---------------------------------------------------------------------------
#define NSTR 72   // 64 + 8 pad (144B rows)
__global__ __launch_bounds__(256) void gemm_nt_k(
    const unsigned short* __restrict__ X,
    const unsigned short* __restrict__ P,
    unsigned short* __restrict__ C) {
  __shared__ unsigned short Al[128 * NSTR];
  __shared__ unsigned short Bl[128 * NSTR];
  int tid = threadIdx.x;
  int wave = tid >> 5, lane = tid & 31;
  int hh = lane >> 4, n16 = lane & 15;
  int z = blockIdx.z;           // b*NH + h
  int b = z >> 4, h = z & 15;
  int gm0 = blockIdx.y * 128;   // s tile
  int gn0 = blockIdx.x * 128;   // bucket tile
  int wm = (wave >> 2) * 64, wn = (wave & 3) * 32;

  // 128x64 tiles: 512 32-byte chunks -> 2 per thread per tile
#pragma unroll
  for (int it = 0; it < 2; it++) {
    int chunk = tid + it * 256;
    int r = chunk >> 2, c = (chunk & 3) * 16;
    cp16B(X + (size_t)(b * SEQ + gm0 + r) * DM + h * HDIM + c,     &Al[r * NSTR + c]);
    cp16B(X + (size_t)(b * SEQ + gm0 + r) * DM + h * HDIM + c + 8, &Al[r * NSTR + c + 8]);
    cp16B(P + (size_t)(gn0 + r) * DM + h * HDIM + c,               &Bl[r * NSTR + c]);
    cp16B(P + (size_t)(gn0 + r) * DM + h * HDIM + c + 8,           &Bl[r * NSTR + c + 8]);
  }
  cp_commit();
  __syncthreads();

  v8f acc[4][2];
  const v8f vzero = {0.f,0.f,0.f,0.f,0.f,0.f,0.f,0.f};
#pragma unroll
  for (int i = 0; i < 4; i++)
#pragma unroll
    for (int j = 0; j < 2; j++) acc[i][j] = vzero;

#pragma unroll
  for (int ks = 0; ks < 2; ks++) {
    BF16Frag af[4], bfg[2];
#pragma unroll
    for (int i = 0; i < 4; i++) {
      const unsigned short* p = &Al[(wm + i * 16 + n16) * NSTR + ks * 32 + hh * 8];
      af[i].q[0] = *(const v4u*)(const void*)p;
      af[i].q[1] = *(const v4u*)(const void*)(p + 16);
    }
#pragma unroll
    for (int j = 0; j < 2; j++) {
      const unsigned short* p = &Bl[(wn + j * 16 + n16) * NSTR + ks * 32 + hh * 8];
      bfg[j].q[0] = *(const v4u*)(const void*)p;
      bfg[j].q[1] = *(const v4u*)(const void*)(p + 16);
    }
#pragma unroll
    for (int i = 0; i < 4; i++)
#pragma unroll
      for (int j = 0; j < 2; j++)
        acc[i][j] = __builtin_amdgcn_wmma_f32_16x16x32_bf16(
            false, af[i].bf, false, bfg[j].bf, (short)0, acc[i][j], false, false);
  }

#pragma unroll
  for (int i = 0; i < 4; i++) {
    int row = gm0 + wm + i * 16 + hh * 8;
#pragma unroll
    for (int j = 0; j < 2; j++) {
      int col = gn0 + wn + j * 16 + n16;
#pragma unroll
      for (int r = 0; r < 8; r++)
        C[((size_t)z * SEQ + row + r) * K2 + col] = f2bf(acc[i][j][r]);
    }
  }
}

// ---------------------------------------------------------------------------
// Flash attention with disentangled relative bias.
// Grid: (SEQ/128, NB*NH). 8 waves; wave w owns query rows [w*16, w*16+16).
// ---------------------------------------------------------------------------
__global__ __launch_bounds__(256) void flash_attn_k(
    const unsigned short* __restrict__ Qb,
    const unsigned short* __restrict__ Kb,
    const unsigned short* __restrict__ Vb,
    const unsigned short* __restrict__ C2P,   // [NB*NH*SEQ, K2]
    const unsigned short* __restrict__ P2C,   // [NB*NH*SEQ, K2]
    const int* __restrict__ relTab,           // [2048]
    unsigned short* __restrict__ Ctx) {       // [NB*SEQ, DM]
  __shared__ unsigned short Ql[128 * NSTR];
  __shared__ unsigned short Kl[64 * NSTR];
  __shared__ unsigned short Vtl[64 * NSTR];
  __shared__ unsigned short Pl[8 * 16 * NSTR];
  __shared__ int tab[2048];

  int tid = threadIdx.x;
  int wave = tid >> 5, lane = tid & 31;
  int hh = lane >> 4, n16 = lane & 15;
  int z = blockIdx.y;  int b = z >> 4, h = z & 15;
  int q0 = blockIdx.x * 128;
  size_t bhBase = (size_t)z * SEQ;

#pragma unroll
  for (int e = 0; e < 8; e++) tab[tid + e * 256] = relTab[tid + e * 256];

  // Q tile 128x64 (2 chunks of 16 bf16 per thread), async path
#pragma unroll
  for (int it = 0; it < 2; it++) {
    int chunk = tid + it * 256;
    int r = chunk >> 2, c = (chunk & 3) * 16;
    const unsigned short* gp = Qb + (size_t)(b * SEQ + q0 + r) * DM + h * HDIM + c;
    cp16B(gp,     &Ql[r * NSTR + c]);
    cp16B(gp + 8, &Ql[r * NSTR + c + 8]);
  }
  cp_commit();
  __syncthreads();

  int wq0 = wave * 16;
  BF16Frag aq[2];
#pragma unroll
  for (int ks = 0; ks < 2; ks++) {
    const unsigned short* p = &Ql[(wq0 + n16) * NSTR + ks * 32 + hh * 8];
    aq[ks].q[0] = *(const v4u*)(const void*)p;
    aq[ks].q[1] = *(const v4u*)(const void*)(p + 16);
  }

  float mrun[8], lrun[8];
  v8f oacc[4];
  const v8f vzero = {0.f,0.f,0.f,0.f,0.f,0.f,0.f,0.f};
#pragma unroll
  for (int r = 0; r < 8; r++) { mrun[r] = -1e30f; lrun[r] = 0.f; }
#pragma unroll
  for (int f = 0; f < 4; f++) oacc[f] = vzero;

  const float inv192 = 0.07216878364870323f;  // 1/sqrt(hd*3)
  const float inv128 = 0.08838834764831845f;  // 1/sqrt(hd*2)

  const int kr4 = tid >> 2, kc4 = (tid & 3) * 16;  // 64x64 tile: 1 chunk/thread

  for (int kb = 0; kb < SEQ; kb += 64) {
    // K tile 64x64 row-major (async), V tile transposed (register-staged)
    {
      const unsigned short* gp = Kb + (size_t)(b * SEQ + kb + kr4) * DM + h * HDIM + kc4;
      cp16B(gp, &Kl[kr4 * NSTR + kc4]);
      cp16B(gp + 8, &Kl[kr4 * NSTR + kc4 + 8]);
    }
    {
      const unsigned short* gp = Vb + (size_t)(b * SEQ + kb + kr4) * DM + h * HDIM + kc4;
      Stage16 t;
      t.q[0] = *(const v4u*)(const void*)gp;
      t.q[1] = *(const v4u*)(const void*)(gp + 8);
#pragma unroll
      for (int e = 0; e < 16; e++) Vtl[(kc4 + e) * NSTR + kr4] = t.us[e];
    }
    cp_commit();
    __syncthreads();

    // scores = Q . K^T (contraction over d)
    v8f sacc[4];
#pragma unroll
    for (int f = 0; f < 4; f++) sacc[f] = vzero;
#pragma unroll
    for (int ks = 0; ks < 2; ks++) {
#pragma unroll
      for (int f = 0; f < 4; f++) {
        BF16Frag bk;
        const unsigned short* p = &Kl[(f * 16 + n16) * NSTR + ks * 32 + hh * 8];
        bk.q[0] = *(const v4u*)(const void*)p;
        bk.q[1] = *(const v4u*)(const void*)(p + 16);
        sacc[f] = __builtin_amdgcn_wmma_f32_16x16x32_bf16(
            false, aq[ks].bf, false, bk.bf, (short)0, sacc[f], false, false);
      }
    }

    // add disentangled biases (L2-resident gathers)
    float sv[4][8];
#pragma unroll
    for (int f = 0; f < 4; f++) {
      int k = kb + f * 16 + n16;
#pragma unroll
      for (int r = 0; r < 8; r++) {
        int s = q0 + wq0 + hh * 8 + r;
        int bi = tab[s - k + 1023];
        float c2 = bf2f(C2P[(bhBase + s) * (size_t)K2 + bi]);
        float p2 = bf2f(P2C[(bhBase + k) * (size_t)K2 + bi]);
        sv[f][r] = sacc[f][r] * inv192 + (c2 + p2) * inv128;
      }
    }

    // online softmax (rows live across 16-lane halves; xor 1,2,4,8)
#pragma unroll
    for (int r = 0; r < 8; r++) {
      float rm = fmaxf(fmaxf(sv[0][r], sv[1][r]), fmaxf(sv[2][r], sv[3][r]));
      rm = fmaxf(rm, __shfl_xor(rm, 1, 32));
      rm = fmaxf(rm, __shfl_xor(rm, 2, 32));
      rm = fmaxf(rm, __shfl_xor(rm, 4, 32));
      rm = fmaxf(rm, __shfl_xor(rm, 8, 32));
      float mnew = fmaxf(mrun[r], rm);
      float scale = __expf(mrun[r] - mnew);
      float rs = 0.f;
#pragma unroll
      for (int f = 0; f < 4; f++) { sv[f][r] = __expf(sv[f][r] - mnew); rs += sv[f][r]; }
      rs += __shfl_xor(rs, 1, 32);
      rs += __shfl_xor(rs, 2, 32);
      rs += __shfl_xor(rs, 4, 32);
      rs += __shfl_xor(rs, 8, 32);
      lrun[r] = lrun[r] * scale + rs;
      mrun[r] = mnew;
#pragma unroll
      for (int f = 0; f < 4; f++) oacc[f][r] *= scale;
    }

    // write P (bf16) to per-wave LDS to re-shape into A-fragments
    unsigned short* pw = &Pl[wave * 16 * NSTR];
#pragma unroll
    for (int f = 0; f < 4; f++)
#pragma unroll
      for (int r = 0; r < 8; r++)
        pw[(hh * 8 + r) * NSTR + f * 16 + n16] = f2bf(sv[f][r]);

    // O += P . V (contraction over 64 keys)
#pragma unroll
    for (int ks = 0; ks < 2; ks++) {
      BF16Frag ap;
      const unsigned short* p = &pw[n16 * NSTR + ks * 32 + hh * 8];
      ap.q[0] = *(const v4u*)(const void*)p;
      ap.q[1] = *(const v4u*)(const void*)(p + 16);
#pragma unroll
      for (int f = 0; f < 4; f++) {
        BF16Frag bv;
        const unsigned short* vp = &Vtl[(f * 16 + n16) * NSTR + ks * 32 + hh * 8];
        bv.q[0] = *(const v4u*)(const void*)vp;
        bv.q[1] = *(const v4u*)(const void*)(vp + 16);
        oacc[f] = __builtin_amdgcn_wmma_f32_16x16x32_bf16(
            false, ap.bf, false, bv.bf, (short)0, oacc[f], false, false);
      }
    }
    __syncthreads();
  }

#pragma unroll
  for (int f = 0; f < 4; f++) {
    int d = h * HDIM + f * 16 + n16;
#pragma unroll
    for (int r = 0; r < 8; r++) {
      int s = q0 + wq0 + hh * 8 + r;
      Ctx[(size_t)(b * SEQ + s) * DM + d] = f2bf(oacc[f][r] / lrun[r]);
    }
  }
}

// ---------------------------------------------------------------------------
// Launch
// ---------------------------------------------------------------------------
extern "C" void kernel_launch(void* const* d_in, const int* in_sizes, int n_in,
                              void* d_out, int out_size, void* d_ws, size_t ws_size,
                              hipStream_t stream) {
  const float* hs   = (const float*)d_in[0];
  const float* Wq   = (const float*)d_in[1];  const float* bq  = (const float*)d_in[2];
  const float* Wk   = (const float*)d_in[3];  const float* bk  = (const float*)d_in[4];
  const float* Wv   = (const float*)d_in[5];  const float* bv  = (const float*)d_in[6];
  const float* Wo   = (const float*)d_in[7];  const float* bo  = (const float*)d_in[8];
  const float* rel  = (const float*)d_in[9];
  const float* lng  = (const float*)d_in[10]; const float* lnb = (const float*)d_in[11];
  const float* Wpk  = (const float*)d_in[12]; const float* bpk = (const float*)d_in[13];
  const float* Wpq  = (const float*)d_in[14]; const float* bpq = (const float*)d_in[15];
  float* out = (float*)d_out;

  char* ws = (char*)d_ws;
  auto alloc = [&](size_t bytes) -> char* {
    char* p = ws; ws += (bytes + 255) & ~(size_t)255; return p;
  };
  const size_t BS = (size_t)NB * SEQ;           // 4096
  unsigned short* hs_bf  = (unsigned short*)alloc(BS * DM * 2);
  unsigned short* wq_bf  = (unsigned short*)alloc((size_t)DM * DM * 2);
  unsigned short* wk_bf  = (unsigned short*)alloc((size_t)DM * DM * 2);
  unsigned short* wv_bf  = (unsigned short*)alloc((size_t)DM * DM * 2);
  unsigned short* wo_bf  = (unsigned short*)alloc((size_t)DM * DM * 2);
  unsigned short* wpk_bf = (unsigned short*)alloc((size_t)DM * DM * 2);
  unsigned short* wpq_bf = (unsigned short*)alloc((size_t)DM * DM * 2);
  unsigned short* re_bf  = (unsigned short*)alloc((size_t)K2 * DM * 2);
  unsigned short* q_bf   = (unsigned short*)alloc(BS * DM * 2);
  unsigned short* k_bf   = (unsigned short*)alloc(BS * DM * 2);
  unsigned short* v_bf   = (unsigned short*)alloc(BS * DM * 2);
  unsigned short* pk_bf  = (unsigned short*)alloc((size_t)K2 * DM * 2);
  unsigned short* pq_bf  = (unsigned short*)alloc((size_t)K2 * DM * 2);
  unsigned short* c2p    = (unsigned short*)alloc((size_t)NB * NH * SEQ * K2 * 2);
  unsigned short* p2c    = (unsigned short*)alloc((size_t)NB * NH * SEQ * K2 * 2);
  unsigned short* ctx_bf = (unsigned short*)alloc(BS * DM * 2);
  int* tab = (int*)alloc(2048 * 4);

  const int nBSD = (int)(BS * DM);         // 4 Mi
  const int nDD  = DM * DM;                // 1 Mi
  cvt_bf16_k<<<(nBSD + 255) / 256, 256, 0, stream>>>(hs,  hs_bf,  nBSD);
  cvt_bf16_k<<<(nDD  + 255) / 256, 256, 0, stream>>>(Wq,  wq_bf,  nDD);
  cvt_bf16_k<<<(nDD  + 255) / 256, 256, 0, stream>>>(Wk,  wk_bf,  nDD);
  cvt_bf16_k<<<(nDD  + 255) / 256, 256, 0, stream>>>(Wv,  wv_bf,  nDD);
  cvt_bf16_k<<<(nDD  + 255) / 256, 256, 0, stream>>>(Wo,  wo_bf,  nDD);
  cvt_bf16_k<<<(nDD  + 255) / 256, 256, 0, stream>>>(Wpk, wpk_bf, nDD);
  cvt_bf16_k<<<(nDD  + 255) / 256, 256, 0, stream>>>(Wpq, wpq_bf, nDD);
  ln_bf16_k<<<K2, 256, 0, stream>>>(rel, lng, lnb, re_bf);
  relidx_k<<<8, 256, 0, stream>>>(tab);

  dim3 gQKV(DM / 128, (int)(BS / 128));   // (8, 32)
  gemm_rm_k<<<gQKV, 256, 0, stream>>>(hs_bf, DM, wq_bf, DM, bq, q_bf, DM,
                                      (int)BS, DM, DM, 0);
  gemm_rm_k<<<gQKV, 256, 0, stream>>>(hs_bf, DM, wk_bf, DM, bk, k_bf, DM,
                                      (int)BS, DM, DM, 0);
  gemm_rm_k<<<gQKV, 256, 0, stream>>>(hs_bf, DM, wv_bf, DM, bv, v_bf, DM,
                                      (int)BS, DM, DM, 0);
  dim3 gPOS(DM / 128, K2 / 128);          // (8, 4)
  gemm_rm_k<<<gPOS, 256, 0, stream>>>(re_bf, DM, wpk_bf, DM, bpk, pk_bf, DM,
                                      K2, DM, DM, 0);
  gemm_rm_k<<<gPOS, 256, 0, stream>>>(re_bf, DM, wpq_bf, DM, bpq, pq_bf, DM,
                                      K2, DM, DM, 0);

  dim3 gNT(K2 / 128, SEQ / 128, NB * NH); // (4, 8, 64)
  gemm_nt_k<<<gNT, 256, 0, stream>>>(q_bf, pk_bf, c2p);
  gemm_nt_k<<<gNT, 256, 0, stream>>>(k_bf, pq_bf, p2c);

  dim3 gFA(SEQ / 128, NB * NH);           // (8, 64)
  flash_attn_k<<<gFA, 256, 0, stream>>>(q_bf, k_bf, v_bf, c2p, p2c, tab, ctx_bf);

  gemm_rm_k<<<gQKV, 256, 0, stream>>>(ctx_bf, DM, wo_bf, DM, bo, out, DM,
                                      (int)BS, DM, DM, 1);
  (void)in_sizes; (void)n_in; (void)out_size; (void)ws_size;
}
